// ConvProjector_6820408066821
// MI455X (gfx1250) — matrix operand order repaired and checked by reading, not compiled, not run on hardware
//
#include <hip/hip_runtime.h>

// ---------------------------------------------------------------------------
// ConvProjector for MI455X (gfx1250, wave32, WMMA + async global->LDS).
//
//  conv3x3(576->256)+ReLU -> conv3x3(256->256)+ReLU  at 32x32  [bf16 WMMA]
//  then (reordered, exact): 1x1 reduce 256->24 at 32x32, THEN adaptive
//  avg-pool upsample 24ch 32x32 -> 687x1024 (window <= 2x1 here).
// ---------------------------------------------------------------------------

typedef __attribute__((ext_vector_type(16))) __bf16 v16bf;
typedef __attribute__((ext_vector_type(8)))  float  v8f;

typedef __attribute__((address_space(3))) unsigned char lds_byte;
typedef __attribute__((address_space(3))) const v16bf   lds_v16bf;

#define IN_C   576
#define MID_C  256
#define OUT_C  24
#define NPIX   1024          // 32*32
#define K1     (IN_C * 9)    // 5184, divisible by 32
#define K2     (MID_C * 9)   // 2304, divisible by 32
#define OUT_H  687
#define OUT_W  1024

__device__ __forceinline__ unsigned short f2bf(float f) {
  union { float f; unsigned u; } v; v.f = f;
  unsigned u = v.u;
  u += 0x7FFFu + ((u >> 16) & 1u);      // round-to-nearest-even
  return (unsigned short)(u >> 16);
}

// --------------------------------------------------------------------------
// Swizzle f32 row-major weights [M][K] into WMMA A-fragment tiles:
// tile (mt,kt) stored as 32 lanes x 16 bf16 contiguous per lane.
// A fragment k order used here (contiguous): lane<16 holds m=mt*16+lane, K offs 0..15;
// lane>=16 holds same rows, K offs 16..31.
// --------------------------------------------------------------------------
__global__ void swizzle_w_kernel(const float* __restrict__ W,
                                 unsigned short* __restrict__ A, int M, int K) {
  int idx = blockIdx.x * blockDim.x + threadIdx.x;
  if (idx >= M * K) return;
  int j    = idx & 15;
  int lane = (idx >> 4) & 31;
  int t    = idx >> 9;                 // mt*Kt + kt
  int Kt   = K >> 5;
  int kt   = t % Kt;
  int mt   = t / Kt;
  int m    = mt * 16 + (lane & 15);
  int ko   = (lane < 16) ? j : 16 + j;
  A[idx] = f2bf(W[(size_t)m * K + kt * 32 + ko]);
}

// --------------------------------------------------------------------------
// im2col of f32 NCHW 32x32 input into WMMA B-fragment tiles:
// tile (kt,nt): lane<16 holds col n=nt*16+lane, K offs 0..15;
// lane>=16 holds same col, K offs 16..31. k = c*9 + (ky*3+kx).
// --------------------------------------------------------------------------
__global__ void im2col_f32_kernel(const float* __restrict__ X,
                                  unsigned short* __restrict__ B, int K) {
  int idx = blockIdx.x * blockDim.x + threadIdx.x;
  if (idx >= K * NPIX) return;
  int j    = idx & 15;
  int lane = (idx >> 4) & 31;
  int t    = idx >> 9;                 // kt*Nt + nt
  int Nt   = NPIX / 16;
  int nt   = t % Nt;
  int kt   = t / Nt;
  int n    = nt * 16 + (lane & 15);
  int k    = kt * 32 + ((lane < 16) ? j : 16 + j);
  int c = k / 9, tap = k % 9;
  int y = (n >> 5) + tap / 3 - 1;
  int x = (n & 31) + tap % 3 - 1;
  float v = 0.0f;
  if ((unsigned)y < 32u && (unsigned)x < 32u) v = X[((size_t)c * 32 + y) * 32 + x];
  B[idx] = f2bf(v);
}

// Same, but source is bf16 activations [C][1024].
__global__ void im2col_bf16_kernel(const unsigned short* __restrict__ H,
                                   unsigned short* __restrict__ B, int K) {
  int idx = blockIdx.x * blockDim.x + threadIdx.x;
  if (idx >= K * NPIX) return;
  int j    = idx & 15;
  int lane = (idx >> 4) & 31;
  int t    = idx >> 9;
  int Nt   = NPIX / 16;
  int nt   = t % Nt;
  int kt   = t / Nt;
  int n    = nt * 16 + (lane & 15);
  int k    = kt * 32 + ((lane < 16) ? j : 16 + j);
  int c = k / 9, tap = k % 9;
  int y = (n >> 5) + tap / 3 - 1;
  int x = (n & 31) + tap % 3 - 1;
  unsigned short v = 0;
  if ((unsigned)y < 32u && (unsigned)x < 32u) v = H[((size_t)c * 32 + y) * 32 + x];
  B[idx] = v;
}

// --------------------------------------------------------------------------
// bf16 WMMA GEMM with async global->LDS double-buffered staging.
// Block = 128 threads = 2x2 waves; macro-tile 64(M) x 64(N); K-step 32.
// Per K-step: block async-copies A chunk (4 tiles x 1KB) + B chunk (4 x 1KB)
// into LDS (ASYNCcnt), waits + barriers, each wave reads its fragments from
// LDS (ds_load_b128) and does 4 x v_wmma_f32_16x16x32_bf16.
// A: [Mt][Kt][32 lanes][16 bf16], B: [Kt][Nt][32 lanes][16 bf16].
// Epilogue: +bias, ReLU, store f32 and/or bf16 row-major [M][N].
// --------------------------------------------------------------------------
__global__ void __launch_bounds__(128)
gemm_bf16_kernel(const unsigned short* __restrict__ A,
                 const unsigned short* __restrict__ Bm,
                 const float* __restrict__ bias,
                 float* __restrict__ outF,
                 unsigned short* __restrict__ outBf,
                 int Ksteps, int N) {
  __shared__ __attribute__((aligned(32))) unsigned char smem[2 * 8192];
  // AS(3) view of smem: ptrtoint of this yields the raw 32-bit LDS byte
  // address (what the async instruction's VDST wants) AND makes smem escape,
  // so the "memory"-clobbering asm is known to write it (keeps ds loads live).
  lds_byte* s3 = (lds_byte*)smem;

  const int tid  = threadIdx.x;
  const int lane = tid & 31;
  const int wave = tid >> 5;          // 0..3
  const int wy   = wave >> 1;         // M half
  const int wx   = wave & 1;          // N half
  const int Nt   = N >> 4;
  const int mtB  = blockIdx.y * 4;    // 4 mt tiles per block
  const int ntB  = blockIdx.x * 4;    // 4 nt tiles per block

  // Copy role: thread `tid` moves 32B of the A chunk and 32B of the B chunk.
  const int      cti  = tid >> 5;         // which tile within chunk (0..3)
  const unsigned coff = (unsigned)((tid & 31) * 32);

  const unsigned long long Abase = (unsigned long long)(size_t)A;
  const unsigned long long Bbase = (unsigned long long)(size_t)Bm;

  auto issue = [&](int kt, int buf) {
    unsigned la = (unsigned)(size_t)(s3 + buf * 8192 + cti * 1024 + coff);
    unsigned lb = la + 4096u;
    unsigned long long ga =
        Abase + ((size_t)(mtB + cti) * (size_t)Ksteps + (size_t)kt) * 1024ull + coff;
    unsigned long long gb =
        Bbase + ((size_t)kt * (size_t)Nt + (size_t)(ntB + cti)) * 1024ull + coff;
    asm volatile(
        "global_load_async_to_lds_b128 %0, %1, off\n\t"
        "global_load_async_to_lds_b128 %0, %1, off offset:16\n\t"
        "global_load_async_to_lds_b128 %2, %3, off\n\t"
        "global_load_async_to_lds_b128 %2, %3, off offset:16"
        :: "v"(la), "v"(ga), "v"(lb), "v"(gb)
        : "memory");
  };

  v8f c00 = {}, c01 = {}, c10 = {}, c11 = {};

  issue(0, 0);
  for (int kt = 0; kt < Ksteps; ++kt) {
    const int buf = kt & 1;
    asm volatile("s_wait_asynccnt 0x0" ::: "memory");  // my copies landed in LDS
    __syncthreads();                                   // everyone's copies landed
    if (kt + 1 < Ksteps) issue(kt + 1, buf ^ 1);       // prefetch next K-step

    lds_byte* sb = s3 + buf * 8192;
    v16bf a0 = *(lds_v16bf*)(sb + (wy * 2 + 0) * 1024 + lane * 32);
    v16bf a1 = *(lds_v16bf*)(sb + (wy * 2 + 1) * 1024 + lane * 32);
    v16bf b0 = *(lds_v16bf*)(sb + 4096 + (wx * 2 + 0) * 1024 + lane * 32);
    v16bf b1 = *(lds_v16bf*)(sb + 4096 + (wx * 2 + 1) * 1024 + lane * 32);

    c00 = __builtin_amdgcn_wmma_f32_16x16x32_bf16(false, a0, false, b0, (short)0, c00, false, false);
    c01 = __builtin_amdgcn_wmma_f32_16x16x32_bf16(false, a0, false, b1, (short)0, c01, false, false);
    c10 = __builtin_amdgcn_wmma_f32_16x16x32_bf16(false, a1, false, b0, (short)0, c10, false, false);
    c11 = __builtin_amdgcn_wmma_f32_16x16x32_bf16(false, a1, false, b1, (short)0, c11, false, false);
  }

  // C/D layout: VGPR r, lanes 0-15 -> M=r, lanes 16-31 -> M=r+8; N=lane&15.
  const int mt0    = mtB + wy * 2;
  const int nt0    = ntB + wx * 2;
  const int nl     = lane & 15;
  const int rowoff = (lane >> 4) * 8;
  auto emit = [&](const v8f& cc, int mt, int nt) {
    #pragma unroll
    for (int r = 0; r < 8; ++r) {
      int m = mt * 16 + rowoff + r;
      int n = nt * 16 + nl;
      float v = cc[r] + bias[m];
      v = v > 0.0f ? v : 0.0f;
      if (outF)  outF [(size_t)m * N + n] = v;
      if (outBf) outBf[(size_t)m * N + n] = f2bf(v);
    }
  };
  emit(c00, mt0,     nt0);
  emit(c01, mt0,     nt0 + 1);
  emit(c10, mt0 + 1, nt0);
  emit(c11, mt0 + 1, nt0 + 1);
}

// 1x1 channel reduce 256->24 at 32x32 (12.6 MFLOP, exact f32).
__global__ void reduce_1x1_kernel(const float* __restrict__ H2,
                                  const float* __restrict__ Wr,
                                  const float* __restrict__ br,
                                  float* __restrict__ P) {
  int idx = blockIdx.x * blockDim.x + threadIdx.x;
  if (idx >= OUT_C * NPIX) return;
  int n = idx & (NPIX - 1);
  int o = idx >> 10;
  float acc = br[o];
  #pragma unroll 4
  for (int c = 0; c < MID_C; ++c)
    acc += Wr[o * MID_C + c] * H2[(size_t)c * NPIX + n];
  P[idx] = acc;
}

// Adaptive avg-pool upsample 24 x 32x32 -> 24 x 687x1024 (streaming write).
// Width windows are exactly 1 (1024 % 32 == 0); height windows are 1 or 2.
__global__ void upsample_kernel(const float* __restrict__ P,
                                float* __restrict__ Out) {
  size_t idx = (size_t)blockIdx.x * blockDim.x + threadIdx.x;
  const size_t total = (size_t)OUT_C * OUT_H * OUT_W;
  if (idx >= total) return;
  int    x = (int)(idx & (OUT_W - 1));
  size_t r = idx >> 10;
  int    y = (int)(r % OUT_H);
  int    o = (int)(r / OUT_H);
  int s = (y * 32) / OUT_H;
  int e = (y * 32 + 32 + OUT_H - 1) / OUT_H;   // ceil((y+1)*32/687)
  const float* row = P + ((size_t)o * 32 + s) * 32 + (x >> 5);
  float v = row[0];
  if (e - s == 2) v = 0.5f * (v + row[32]);
  Out[idx] = v;
}

// ---------------------------------------------------------------------------
extern "C" void kernel_launch(void* const* d_in, const int* in_sizes, int n_in,
                              void* d_out, int out_size, void* d_ws, size_t ws_size,
                              hipStream_t stream) {
  const float* x  = (const float*)d_in[0];
  const float* w1 = (const float*)d_in[1];
  const float* b1 = (const float*)d_in[2];
  const float* w2 = (const float*)d_in[3];
  const float* b2 = (const float*)d_in[4];
  const float* wr = (const float*)d_in[5];
  const float* br = (const float*)d_in[6];
  float* out = (float*)d_out;

  char* ws = (char*)d_ws;
  size_t off = 0;
  auto alloc = [&](size_t bytes) {
    void* p = ws + off;
    off = (off + bytes + 255) & ~(size_t)255;
    return p;
  };
  unsigned short* w1b = (unsigned short*)alloc((size_t)MID_C * K1 * 2);
  unsigned short* w2b = (unsigned short*)alloc((size_t)MID_C * K2 * 2);
  unsigned short* im1 = (unsigned short*)alloc((size_t)K1 * NPIX * 2);
  unsigned short* im2 = (unsigned short*)alloc((size_t)K2 * NPIX * 2);
  unsigned short* h1b = (unsigned short*)alloc((size_t)MID_C * NPIX * 2);
  float*          h2f = (float*)alloc((size_t)MID_C * NPIX * 4);
  float*          p24 = (float*)alloc((size_t)OUT_C * NPIX * 4);

  int t;
  t = MID_C * K1;
  swizzle_w_kernel<<<(t + 255) / 256, 256, 0, stream>>>(w1, w1b, MID_C, K1);
  t = MID_C * K2;
  swizzle_w_kernel<<<(t + 255) / 256, 256, 0, stream>>>(w2, w2b, MID_C, K2);
  t = K1 * NPIX;
  im2col_f32_kernel<<<(t + 255) / 256, 256, 0, stream>>>(x, im1, K1);

  // conv1: M=256, N=1024, K=5184  -> 64x64 macro-tiles: grid (16, 4)
  gemm_bf16_kernel<<<dim3(NPIX / 64, MID_C / 64), 128, 0, stream>>>(
      w1b, im1, b1, nullptr, h1b, K1 / 32, NPIX);

  t = K2 * NPIX;
  im2col_bf16_kernel<<<(t + 255) / 256, 256, 0, stream>>>(h1b, im2, K2);

  // conv2: M=256, N=1024, K=2304
  gemm_bf16_kernel<<<dim3(NPIX / 64, MID_C / 64), 128, 0, stream>>>(
      w2b, im2, b2, h2f, nullptr, K2 / 32, NPIX);

  t = OUT_C * NPIX;
  reduce_1x1_kernel<<<(t + 255) / 256, 256, 0, stream>>>(h2f, wr, br, p24);

  size_t total = (size_t)OUT_C * OUT_H * OUT_W;
  upsample_kernel<<<(unsigned)((total + 255) / 256), 256, 0, stream>>>(p24, out);
}
